// BaselineMAMBAFlow_31679678775741
// MI455X (gfx1250) — matrix operand order repaired and checked by reading, not compiled, not run on hardware
//
#include <hip/hip_runtime.h>
#include <hip/hip_bf16.h>

typedef __attribute__((ext_vector_type(16))) _Float16 v16h;
typedef __attribute__((ext_vector_type(8)))  _Float16 v8h;
typedef __attribute__((ext_vector_type(8)))  float    v8f;
typedef __attribute__((ext_vector_type(4)))  unsigned int u32x4;
typedef __attribute__((ext_vector_type(8)))  int i32x8;
typedef __attribute__((ext_vector_type(4)))  int i32x4;

#define WMMA_F32_F16(a, b, c) \
  __builtin_amdgcn_wmma_f32_16x16x32_f16(false, (a), false, (b), (short)0, (c), false, false)

#if defined(__has_builtin)
#if __has_builtin(__builtin_amdgcn_tensor_load_to_lds) && __has_builtin(__builtin_amdgcn_s_wait_tensorcnt)
#define HAVE_TDM 1
#endif
#endif

static __device__ __forceinline__ v16h combine(v8h lo, v8h hi) {
  return __builtin_shufflevector(lo, hi, 0, 1, 2, 3, 4, 5, 6, 7, 8, 9, 10, 11, 12, 13, 14, 15);
}

static __device__ __forceinline__ v16h cvt16(float4 a, float4 b, float4 c, float4 d) {
  v16h r;
  r[0] = (_Float16)a.x;  r[1] = (_Float16)a.y;  r[2] = (_Float16)a.z;  r[3] = (_Float16)a.w;
  r[4] = (_Float16)b.x;  r[5] = (_Float16)b.y;  r[6] = (_Float16)b.z;  r[7] = (_Float16)b.w;
  r[8] = (_Float16)c.x;  r[9] = (_Float16)c.y;  r[10] = (_Float16)c.z; r[11] = (_Float16)c.w;
  r[12] = (_Float16)d.x; r[13] = (_Float16)d.y; r[14] = (_Float16)d.z; r[15] = (_Float16)d.w;
  return r;
}

#ifdef HAVE_TDM
// TDM: DMA a 2-D f32 tile (32 cols x 64 rows, row stride ld elems) from global
// into LDS at lds_byte_off, padding 4 DWORDs every 32 DWORDs -> 36-float rows.
// D# layout per CDNA5 ISA ch.8 (group0 128b, group1 256b; groups 2/3 unused).
// clang-23 toolchain signature: (u32x4, i32x8, i32x4, i32x4, i32x8, i32 cpol).
static __device__ __forceinline__ void tdm_load_tile_f32(
    const float* gsrc, unsigned lds_byte_off, int ld)
{
  unsigned long long ga = (unsigned long long)(uintptr_t)gsrc;
  u32x4 g0;
  g0[0] = 1u;                                                  // count=1 (valid), user mode
  g0[1] = lds_byte_off;                                        // lds_addr (bytes)
  g0[2] = (unsigned)(ga & 0xffffffffu);                        // global_addr[31:0]
  g0[3] = (unsigned)((ga >> 32) & 0x01ffffffu) | 0x80000000u;  // global_addr[56:32] | type=2
  i32x8 g1;
  g1[0] = (2 << 16) | (1 << 20) | (4 << 22) | (3 << 25);       // data_size=4B, pad_en,
                                                               // pad every 32 DW, pad 4 DW
  g1[1] = (int)(32u << 16);                                    // tensor_dim0 = 32
  g1[2] = (int)(64u << 16);                                    // tensor_dim1 = 64
  g1[3] = (int)(32u << 16);                                    // tile_dim0 = 32
  g1[4] = 64;                                                  // tile_dim1 = 64, tile_dim2 = 0
  g1[5] = ld;                                                  // tensor_dim0_stride (elems)
  g1[6] = 0;
  g1[7] = 0;
  i32x4 z4 = {};
  i32x8 z8 = {};
  __builtin_amdgcn_tensor_load_to_lds(g0, g1, z4, z4, z8, 0);
}
#endif

// ---------------------------------------------------------------------------
// Embedding: fourier features -> input proj + time embedding -> seq (B,2048,256)
// ---------------------------------------------------------------------------
__global__ __launch_bounds__(256) void embed_kernel(
    const float* __restrict__ sc, const float* __restrict__ sv,
    const float* __restrict__ qc, const float* __restrict__ tarr,
    const float* __restrict__ noise, const float* __restrict__ Bf,
    const float* __restrict__ Wq, const float* __restrict__ bq,
    const float* __restrict__ Wi, const float* __restrict__ bi,
    float* __restrict__ seq)
{
  __shared__ float s33[33];
  const int tokg = blockIdx.x;          // 0 .. B*2048-1
  const int b = tokg >> 11;
  const int n = tokg & 2047;
  const int d = threadIdx.x;
  const bool isq = (n >= 1024);
  const int nn = isq ? (n - 1024) : n;
  const float* coords = isq ? qc : sc;
  const float* val    = isq ? noise : sv;
  const float* W      = isq ? Wq : Wi;
  const float* bb     = isq ? bq : bi;
  if (d < 16) {
    const long long ci = ((long long)(b * 1024 + nn)) * 2;
    float x = coords[ci + 0], y = coords[ci + 1];
    float p = x * Bf[d] + y * Bf[16 + d];
    s33[d] = sinf(p);
    s33[16 + d] = cosf(p);
  }
  if (d == 32) s33[32] = val[(long long)(b * 1024 + nn)];
  __syncthreads();
  float acc = bb[d];
  for (int j = 0; j < 33; ++j) acc += W[d * 33 + j] * s33[j];
  const float t = tarr[b];
  float te;
  if (d < 128) te = sinf(t * __expf((float)d * (-9.210340372f / 127.f)));
  else         te = cosf(t * __expf((float)(d - 128) * (-9.210340372f / 127.f)));
  seq[(long long)tokg * 256 + d] = acc + te;
}

// ---------------------------------------------------------------------------
// LayerNorm + Bu = xn @ Bw^T   (one block = one token)
// ---------------------------------------------------------------------------
__global__ __launch_bounds__(256) void ln_bu_kernel(
    const float* __restrict__ seq, const float* __restrict__ g,
    const float* __restrict__ beta, const float* __restrict__ Bw,
    float* __restrict__ xn, float* __restrict__ Bu)
{
  __shared__ float red[256];
  __shared__ alignas(16) float sxn[256];
  const long long tok = blockIdx.x;
  const int d = threadIdx.x;
  const float x = seq[tok * 256 + d];
  red[d] = x;
  __syncthreads();
  for (int s = 128; s > 0; s >>= 1) { if (d < s) red[d] += red[d + s]; __syncthreads(); }
  const float mu = red[0] * (1.f / 256.f);
  __syncthreads();
  const float diff = x - mu;
  red[d] = diff * diff;
  __syncthreads();
  for (int s = 128; s > 0; s >>= 1) { if (d < s) red[d] += red[d + s]; __syncthreads(); }
  const float var = red[0] * (1.f / 256.f);
  const float v = diff * rsqrtf(var + 1e-5f) * g[d] + beta[d];
  sxn[d] = v;
  xn[tok * 256 + d] = v;
  __syncthreads();
  if (d < 8) {
    float acc = 0.f;
    const float4* bw = (const float4*)(Bw + d * 256);
    const float4* sp = (const float4*)sxn;
    for (int j = 0; j < 64; ++j) {
      float4 a = sp[j], w4 = bw[j];
      acc += a.x * w4.x + a.y * w4.y + a.z * w4.z + a.w * w4.w;
    }
    Bu[tok * 8 + d] = acc;
  }
}

// ---------------------------------------------------------------------------
// Sequential SSM scan over N=2048 : 128 independent (b,s) chains
// ---------------------------------------------------------------------------
__global__ void scan_kernel(const float* __restrict__ A_log,
                            const float* __restrict__ Bu, float* __restrict__ hs)
{
  const int t = threadIdx.x;
  if (t >= 128) return;
  const int b = t >> 3, s = t & 7;
  float a = __expf(A_log[s]);
  a = fminf(fmaxf(a, 1e-8f), 10.f);
  const float abar = __expf(-a / 2048.f);
  float h = 0.f;
  const float* bu = Bu + (long long)b * 2048 * 8 + s;
  float* out = hs + (long long)b * 2048 * 8 + s;
  for (int n = 0; n < 2048; ++n) {
    h = abar * h + bu[n * 8];
    out[n * 8] = h;
  }
}

// ---------------------------------------------------------------------------
// seq += gate * (hs @ Cw^T + Dp * xn)
// ---------------------------------------------------------------------------
__global__ __launch_bounds__(256) void ssm_update_kernel(
    float* __restrict__ seq, const float* __restrict__ xn,
    const float* __restrict__ gate, const float* __restrict__ hs,
    const float* __restrict__ Cw, const float* __restrict__ Dp)
{
  __shared__ float h8[8];
  const long long tok = blockIdx.x;
  const int d = threadIdx.x;
  if (d < 8) h8[d] = hs[tok * 8 + d];
  __syncthreads();
  float y = Dp[d] * xn[tok * 256 + d];
  const float* cw = Cw + d * 8;
#pragma unroll
  for (int s = 0; s < 8; ++s) y += h8[s] * cw[s];
  seq[tok * 256 + d] += gate[tok * 256 + d] * y;
}

// ---------------------------------------------------------------------------
// TDM-fed, double-buffered WMMA GEMM: C = act(A[M,K] @ W[N,K]^T + bias[N])
// block = 256 threads (8 waves), block tile 64x64, K-step 32.
// Wave 0 issues tensor_load_to_lds for the NEXT K-step's A/W tiles, waits
// s_wait_tensorcnt<=2 (in-order completion => current tiles resident), then
// the block barrier releases the compute waves. Tiles are f32 in LDS with a
// TDM-generated 36-float row stride; fragments = 4x ds_load_b128 + cvt.
// ACT: 0=none, 1=sigmoid, 2=relu
// ---------------------------------------------------------------------------
template <int ACT>
__global__ __launch_bounds__(256) void gemm_wmma(
    const float* __restrict__ A, int lda, long long strideA,
    const float* __restrict__ W, int ldw,
    const float* __restrict__ bias,
    float* __restrict__ C, int ldc, long long strideC,
    int M, int N, int K)
{
  __shared__ alignas(16) float Asf[2][64][36];   // [buf][m][k] (TDM-padded rows)
  __shared__ alignas(16) float Bsf[2][64][36];   // [buf][n][k]
  const int tid  = threadIdx.x;
  const int lane = tid & 31;
  const int wave = tid >> 5;
  const int m0 = blockIdx.y * 64;
  const int n0 = blockIdx.x * 64;
  const float* Ab = A + (long long)blockIdx.z * strideA;
  float* Cb = C + (long long)blockIdx.z * strideC;

  const int mt  = wave >> 1;        // 0..3  (M sub-tile)
  const int nt0 = (wave & 1) * 2;   // 0 or 2 (first of two N sub-tiles)

  v8f acc0 = {}; v8f acc1 = {};

  const int m   = mt * 16 + (lane & 15);
  const int abk = (lane < 16) ? 0 : 8;      // ISA 16-bit A-layout K base
  const int kbb = (lane < 16) ? 0 : 16;     // K base, contiguous order k = 16h + s (A and B)
  const int nc0 = nt0 * 16 + (lane & 15);
  const int nc1 = (nt0 + 1) * 16 + (lane & 15);
  const int ntiles = K >> 5;

#ifdef HAVE_TDM
  if (wave == 0) {
    tdm_load_tile_f32(Ab + (long long)m0 * lda,
                      (unsigned)(uintptr_t)&Asf[0][0][0], lda);
    tdm_load_tile_f32(W + (long long)n0 * ldw,
                      (unsigned)(uintptr_t)&Bsf[0][0][0], ldw);
  }
#endif

  for (int it = 0; it < ntiles; ++it) {
    const int cur = it & 1;
#ifdef HAVE_TDM
    if (wave == 0) {
      if (it + 1 < ntiles) {
        const int kn = (it + 1) << 5;
        tdm_load_tile_f32(Ab + (long long)m0 * lda + kn,
                          (unsigned)(uintptr_t)&Asf[cur ^ 1][0][0], lda);
        tdm_load_tile_f32(W + (long long)n0 * ldw + kn,
                          (unsigned)(uintptr_t)&Bsf[cur ^ 1][0][0], ldw);
        __builtin_amdgcn_s_wait_tensorcnt((short)2);
      } else {
        __builtin_amdgcn_s_wait_tensorcnt((short)0);
      }
    }
#else
    {
      const int srow = tid >> 2;              // 0..63
      const int scol = (tid & 3) * 8;         // 0,8,16,24
      const int k0 = it << 5;
      const float4* a4 = (const float4*)(Ab + (long long)(m0 + srow) * lda + k0 + scol);
      const float4* w4 = (const float4*)(W + (long long)(n0 + srow) * ldw + k0 + scol);
      *(float4*)&Asf[cur][srow][scol]     = a4[0];
      *(float4*)&Asf[cur][srow][scol + 4] = a4[1];
      *(float4*)&Bsf[cur][srow][scol]     = w4[0];
      *(float4*)&Bsf[cur][srow][scol + 4] = w4[1];
    }
#endif
    __syncthreads();

    const float4* ar = (const float4*)&Asf[cur][m][0];
    v16h afrag = cvt16(ar[(kbb >> 2) + 0], ar[(kbb >> 2) + 1],
                       ar[(kbb >> 2) + 2], ar[(kbb >> 2) + 3]);  // K {kbb..kbb+15}
    const float4* b0 = (const float4*)&Bsf[cur][nc0][0];
    const float4* b1 = (const float4*)&Bsf[cur][nc1][0];
    v16h bfrag0 = cvt16(b0[(kbb >> 2) + 0], b0[(kbb >> 2) + 1],
                        b0[(kbb >> 2) + 2], b0[(kbb >> 2) + 3]); // K {kbb..kbb+15}
    v16h bfrag1 = cvt16(b1[(kbb >> 2) + 0], b1[(kbb >> 2) + 1],
                        b1[(kbb >> 2) + 2], b1[(kbb >> 2) + 3]);

    acc0 = WMMA_F32_F16(afrag, bfrag0, acc0);
    acc1 = WMMA_F32_F16(afrag, bfrag1, acc1);
    __syncthreads();
  }

  const int rowbase = (lane < 16) ? 0 : 8;
  const int col = lane & 15;
#pragma unroll
  for (int r = 0; r < 8; ++r) {
    const int gm = m0 + mt * 16 + rowbase + r;
    {
      const int gn = n0 + nt0 * 16 + col;
      float v = acc0[r] + bias[gn];
      if (ACT == 1) v = __builtin_amdgcn_rcpf(1.f + __expf(-v));
      else if (ACT == 2) v = fmaxf(v, 0.f);
      Cb[(long long)gm * ldc + gn] = v;
    }
    {
      const int gn = n0 + (nt0 + 1) * 16 + col;
      float v = acc1[r] + bias[gn];
      if (ACT == 1) v = __builtin_amdgcn_rcpf(1.f + __expf(-v));
      else if (ACT == 2) v = fmaxf(v, 0.f);
      Cb[(long long)gm * ldc + gn] = v;
    }
  }
}

// ---------------------------------------------------------------------------
// Flash attention: block = 4 waves, all on the same (b,h); wave w owns q rows
// [q0, q0+16). 32 keys per iteration staged cooperatively in LDS (K as
// [key][d], V transposed to [d][key]) so all WMMA fragments are b128 LDS
// loads. Online softmax in C-layout; P repacked through per-wave LDS tile.
// ---------------------------------------------------------------------------
__global__ __launch_bounds__(128) void flash_attn(
    const float* __restrict__ Q, const float* __restrict__ Kb,
    const float* __restrict__ Vb, float* __restrict__ O)
{
  __shared__ alignas(16) _Float16 Ks[32][72];      // [key][d]  (64 d + pad)
  __shared__ alignas(16) _Float16 Vs[64][40];      // [d][key]  (32 key + pad)
  __shared__ alignas(16) _Float16 pl[4][16][40];   // per-wave P tile [row][key]
  const int lane = threadIdx.x & 31;
  const int wave = threadIdx.x >> 5;
  const int h = blockIdx.y, b = blockIdx.z;
  const int q0  = (blockIdx.x * 4 + wave) * 16;
  const int hd0 = h * 64;
  const int m = lane & 15;
  const int half16 = (lane < 16) ? 0 : 1;
  const int abk = half16 ? 8 : 0;     // A-layout K base
  const int kb16 = half16 * 16;       // B-layout K base

  v16h qa[2];
  {
    const float* qp = Q + ((long long)(b * 1024 + q0 + m) * 256) + hd0;
#pragma unroll
    for (int c = 0; c < 2; ++c)
#pragma unroll
      for (int e = 0; e < 16; ++e) {
        const int kk = c * 32 + kb16 + e;
        qa[c][e] = (_Float16)qp[kk];
      }
  }
  float mrow[8], lrow[8];
#pragma unroll
  for (int r = 0; r < 8; ++r) { mrow[r] = -1e30f; lrow[r] = 0.f; }
  v8f o0 = {}, o1 = {}, o2 = {}, o3 = {};

  const int skey = threadIdx.x >> 2;          // 0..31
  const int sd0  = (threadIdx.x & 3) * 16;    // 0,16,32,48

  for (int kt = 0; kt < 1024; kt += 32) {
    {
      const float* kp = Kb + ((long long)(b * 1024 + kt + skey) * 256) + hd0 + sd0;
      const float* vp = Vb + ((long long)(b * 1024 + kt + skey) * 256) + hd0 + sd0;
      v8h h0, h1;
#pragma unroll
      for (int i = 0; i < 8; ++i) { h0[i] = (_Float16)kp[i]; h1[i] = (_Float16)kp[8 + i]; }
      *(v8h*)&Ks[skey][sd0] = h0;
      *(v8h*)&Ks[skey][sd0 + 8] = h1;
#pragma unroll
      for (int i = 0; i < 16; ++i) Vs[sd0 + i][skey] = (_Float16)vp[i];
    }
    __syncthreads();

    v8f s0 = {}, s1 = {};
    const v8h* kr0 = (const v8h*)&Ks[m][0];
    const v8h* kr1 = (const v8h*)&Ks[16 + m][0];
#pragma unroll
    for (int c = 0; c < 2; ++c) {
      const int base = (c * 32 + kb16) >> 3;   // 0,2,4,6
      v16h kf0 = combine(kr0[base], kr0[base + 1]);
      v16h kf1 = combine(kr1[base], kr1[base + 1]);
      s0 = WMMA_F32_F16(qa[c], kf0, s0);
      s1 = WMMA_F32_F16(qa[c], kf1, s1);
    }

    float corr[8];
#pragma unroll
    for (int r = 0; r < 8; ++r) {
      const float a0 = s0[r] * 0.125f, a1 = s1[r] * 0.125f;  // 1/sqrt(64)
      float mx = fmaxf(a0, a1);
#pragma unroll
      for (int dd = 1; dd < 16; dd <<= 1) mx = fmaxf(mx, __shfl_xor(mx, dd, 32));
      const float mn = fmaxf(mrow[r], mx);
      const float f  = __expf(mrow[r] - mn);
      const float p0 = __expf(a0 - mn), p1 = __expf(a1 - mn);
      float rs = p0 + p1;
#pragma unroll
      for (int dd = 1; dd < 16; dd <<= 1) rs += __shfl_xor(rs, dd, 32);
      lrow[r] = lrow[r] * f + rs;
      mrow[r] = mn;
      corr[r] = f;
      const int row = r + half16 * 8;
      pl[wave][row][m]      = (_Float16)p0;
      pl[wave][row][16 + m] = (_Float16)p1;
    }
#pragma unroll
    for (int r = 0; r < 8; ++r) {
      o0[r] *= corr[r]; o1[r] *= corr[r]; o2[r] *= corr[r]; o3[r] *= corr[r];
    }

    const v8h* prow = (const v8h*)&pl[wave][m][0];
    v16h pa = combine(prow[kb16 >> 3], prow[(kb16 >> 3) + 1]);

    {
      const v8h* vr0 = (const v8h*)&Vs[0 * 16 + m][0];
      const v8h* vr1 = (const v8h*)&Vs[1 * 16 + m][0];
      const v8h* vr2 = (const v8h*)&Vs[2 * 16 + m][0];
      const v8h* vr3 = (const v8h*)&Vs[3 * 16 + m][0];
      const int base = kb16 >> 3;                 // 0 or 2
      o0 = WMMA_F32_F16(pa, combine(vr0[base], vr0[base + 1]), o0);
      o1 = WMMA_F32_F16(pa, combine(vr1[base], vr1[base + 1]), o1);
      o2 = WMMA_F32_F16(pa, combine(vr2[base], vr2[base + 1]), o2);
      o3 = WMMA_F32_F16(pa, combine(vr3[base], vr3[base + 1]), o3);
    }
    __syncthreads();
  }

#pragma unroll
  for (int r = 0; r < 8; ++r) {
    const float inv = __builtin_amdgcn_rcpf(lrow[r]);
    const int gr = q0 + r + half16 * 8;
    float* dst = O + ((long long)(b * 1024 + gr) * 256) + hd0;
    dst[0 * 16 + m] = o0[r] * inv;
    dst[1 * 16 + m] = o1[r] * inv;
    dst[2 * 16 + m] = o2[r] * inv;
    dst[3 * 16 + m] = o3[r] * inv;
  }
}

// ---------------------------------------------------------------------------
// dec2: out[tok] = h1[tok,:] . w + b
// ---------------------------------------------------------------------------
__global__ __launch_bounds__(256) void dec2_kernel(
    const float* __restrict__ h1, const float* __restrict__ w,
    const float* __restrict__ b2, float* __restrict__ out)
{
  __shared__ float red[256];
  const long long tok = blockIdx.x;
  const int d = threadIdx.x;
  red[d] = h1[tok * 256 + d] * w[d];
  __syncthreads();
  for (int s = 128; s > 0; s >>= 1) { if (d < s) red[d] += red[d + s]; __syncthreads(); }
  if (d == 0) out[tok] = red[0] + b2[0];
}

// ---------------------------------------------------------------------------
extern "C" void kernel_launch(void* const* d_in, const int* in_sizes, int n_in,
                              void* d_out, int out_size, void* d_ws, size_t ws_size,
                              hipStream_t stream)
{
  (void)in_sizes; (void)n_in; (void)out_size; (void)ws_size;
  const float* sc     = (const float*)d_in[0];
  const float* sv     = (const float*)d_in[1];
  const float* qc     = (const float*)d_in[2];
  const float* t      = (const float*)d_in[3];
  const float* noise  = (const float*)d_in[4];
  const float* Bf     = (const float*)d_in[5];
  const float* Wq     = (const float*)d_in[6];
  const float* bq     = (const float*)d_in[7];
  const float* Wi     = (const float*)d_in[8];
  const float* bi     = (const float*)d_in[9];
  const float* A_log  = (const float*)d_in[10];
  const float* Bw     = (const float*)d_in[11];
  const float* Cw     = (const float*)d_in[12];
  const float* Dp     = (const float*)d_in[13];
  const float* ln_g   = (const float*)d_in[14];
  const float* ln_b   = (const float*)d_in[15];
  const float* gate_w = (const float*)d_in[16];
  const float* gate_b = (const float*)d_in[17];
  const float* ipw    = (const float*)d_in[18];
  const float* ipb    = (const float*)d_in[19];
  const float* out_w  = (const float*)d_in[20];
  const float* out_b  = (const float*)d_in[21];
  const float* dw1    = (const float*)d_in[22];
  const float* db1    = (const float*)d_in[23];
  const float* dw2    = (const float*)d_in[24];
  const float* db2    = (const float*)d_in[25];
  float* outp = (float*)d_out;

  // workspace layout (floats); ~98 MB total, attention phase reuses regions
  float* ws   = (float*)d_ws;
  float* seq  = ws;                 // (B,2048,256)  32 MB
  float* xn   = ws + 8388608;       // (B,2048,256)  32 MB
  float* gate = ws + 16777216;      // (B,2048,256)  32 MB
  float* Bu   = ws + 25165824;      // (B,2048,8)     1 MB
  float* hsb  = ws + 25427968;      // (B,2048,8)     1 MB
  float* qb   = xn;                 // (B,1024,256)  (xn free after layers)
  float* kb   = xn + 4194304;       // (B,1024,256)
  float* vb   = gate;               // (B,1024,256)  (gate free after layers)
  float* ob   = gate + 4194304;     // (B,1024,256)
  float* att  = seq;                // (B,1024,256)  (seq free after qkv)
  float* h1   = seq + 4194304;      // (B,1024,256)

  embed_kernel<<<dim3(32768), 256, 0, stream>>>(sc, sv, qc, t, noise, Bf, Wq, bq, Wi, bi, seq);

  for (int l = 0; l < 4; ++l) {
    ln_bu_kernel<<<dim3(32768), 256, 0, stream>>>(seq, ln_g + l * 256, ln_b + l * 256,
                                                  Bw + l * 8 * 256, xn, Bu);
    gemm_wmma<1><<<dim3(4, 512, 1), 256, 0, stream>>>(xn, 256, 0,
                                                      gate_w + l * 65536, 256, gate_b + l * 256,
                                                      gate, 256, 0, 32768, 256, 256);
    scan_kernel<<<dim3(1), 128, 0, stream>>>(A_log + l * 8, Bu, hsb);
    ssm_update_kernel<<<dim3(32768), 256, 0, stream>>>(seq, xn, gate, hsb,
                                                       Cw + l * 2048, Dp + l * 256);
  }

  // q from query half, k/v from input half (batched over B)
  gemm_wmma<0><<<dim3(4, 16, 16), 256, 0, stream>>>(seq + 1024 * 256, 256, 2048 * 256,
                                                    ipw, 256, ipb,
                                                    qb, 256, 1024 * 256, 1024, 256, 256);
  gemm_wmma<0><<<dim3(4, 16, 16), 256, 0, stream>>>(seq, 256, 2048 * 256,
                                                    ipw + 65536, 256, ipb + 256,
                                                    kb, 256, 1024 * 256, 1024, 256, 256);
  gemm_wmma<0><<<dim3(4, 16, 16), 256, 0, stream>>>(seq, 256, 2048 * 256,
                                                    ipw + 131072, 256, ipb + 512,
                                                    vb, 256, 1024 * 256, 1024, 256, 256);

  flash_attn<<<dim3(16, 4, 16), 128, 0, stream>>>(qb, kb, vb, ob);

  gemm_wmma<0><<<dim3(4, 256, 1), 256, 0, stream>>>(ob, 256, 0, out_w, 256, out_b,
                                                    att, 256, 0, 16384, 256, 256);
  gemm_wmma<2><<<dim3(4, 256, 1), 256, 0, stream>>>(att, 256, 0, dw1, 256, db1,
                                                    h1, 256, 0, 16384, 256, 256);
  dec2_kernel<<<dim3(16384), 256, 0, stream>>>(h1, dw2, db2, outp);
}
